// ModernNCA_86715389706322
// MI455X (gfx1250) — matrix operand order repaired and checked by reading, not compiled, not run on hardware
//
#include <hip/hip_runtime.h>
#include <hip/hip_bf16.h>

// ---------------- problem constants ----------------
#define N_ROWS   2048
#define M_CAND   100000
#define D_IN     256
#define DIM      512
#define D_OUT    10
#define EPSV     1e-7f

#define NCHUNK   50                  // M split into 50 chunks of 2000
#define CHUNK    (M_CAND / NCHUNK)   // 2000
#define TILES    (CHUNK / 16)        // 125 candidate tiles per chunk
#define RPB      128                 // rows per block in k_dist (8 waves x 16)
#define RBLKS    (N_ROWS / RPB)      // 16
#define BSTRIDE  520                 // padded LDS row stride (bank-conflict free)

// ---------------- feature detection ----------------
#if defined(__has_builtin)
# if __has_builtin(__builtin_amdgcn_tensor_load_to_lds) && __has_builtin(__builtin_amdgcn_s_wait_tensorcnt)
#  define USE_TDM 1
# endif
# if __has_builtin(__builtin_amdgcn_cvt_pk_bf16_f32)
#  define USE_CVT_PK 1
# endif
#endif
#ifndef USE_TDM
# define USE_TDM 0
#endif
#ifndef USE_CVT_PK
# define USE_CVT_PK 0
#endif

// ---------------- workspace layout (bytes, 256-aligned) ----------------
static constexpr size_t OFF_WT   = 0;                         // 512x256 bf16 (W transposed)
static constexpr size_t SZ_WT    = (size_t)DIM * D_IN * 2;
static constexpr size_t OFF_XE   = OFF_WT + SZ_WT;            // 2048x512 bf16
static constexpr size_t SZ_XE    = (size_t)N_ROWS * DIM * 2;
static constexpr size_t OFF_CE   = OFF_XE + SZ_XE;            // 100000x512 bf16
static constexpr size_t SZ_CE    = (size_t)M_CAND * DIM * 2;
static constexpr size_t OFF_X2   = OFF_CE + SZ_CE;            // 2048 f32
static constexpr size_t SZ_X2    = (size_t)N_ROWS * 4;
static constexpr size_t OFF_C2   = OFF_X2 + SZ_X2;            // 100000 f32
static constexpr size_t SZ_C2    = ((size_t)M_CAND * 4 + 255) & ~(size_t)255;
static constexpr size_t OFF_PZ   = OFF_C2 + SZ_C2;            // [NCHUNK][2048] f32
static constexpr size_t SZ_PZ    = (size_t)NCHUNK * N_ROWS * 4;
static constexpr size_t OFF_PACC = OFF_PZ + SZ_PZ;            // [NCHUNK][2048][10] f32

// ---------------- types ----------------
typedef __bf16 v16bf __attribute__((ext_vector_type(16)));
typedef __bf16 v2bf  __attribute__((ext_vector_type(2)));
typedef float  v8f   __attribute__((ext_vector_type(8)));
typedef unsigned int u32x4 __attribute__((ext_vector_type(4)));
typedef int          i32x8 __attribute__((ext_vector_type(8)));
typedef int          i32x4 __attribute__((ext_vector_type(4)));

union BfVec {
    v16bf          v;
    uint4          q[2];
    unsigned int   u[8];
    unsigned short h[16];
};

// Native bf16 conversion: clang lowers this to the hardware cvt on gfx1250.
__device__ __forceinline__ unsigned short f32_to_bf16(float f) {
    __bf16 h = (__bf16)f;
    unsigned short u;
    __builtin_memcpy(&u, &h, 2);
    return u;
}

#if USE_CVT_PK
__device__ __forceinline__ unsigned int pk_bf16(float a, float b) {
    auto r = __builtin_amdgcn_cvt_pk_bf16_f32(a, b);   // single v_cvt_pk_bf16_f32
    unsigned int u;
    __builtin_memcpy(&u, &r, 4);
    return u;
}
#else
__device__ __forceinline__ unsigned int pk_bf16(float a, float b) {
    v2bf v;
    v[0] = (__bf16)a;          // native casts -> hardware v_cvt (pairable to pk form)
    v[1] = (__bf16)b;
    unsigned int u;
    __builtin_memcpy(&u, &v, 4);
    return u;
}
#endif

__device__ __forceinline__ float bf16_lo(unsigned int u) { return __uint_as_float(u << 16); }
__device__ __forceinline__ float bf16_hi(unsigned int u) { return __uint_as_float(u & 0xffff0000u); }

__device__ __forceinline__ v8f wmma_bf16(v16bf a, v16bf b, v8f c) {
    return __builtin_amdgcn_wmma_f32_16x16x32_bf16(false, a, false, b, (short)0, c, false, false);
}

#if USE_TDM
// Issue one TDM load: 16 rows x 512 bf16 tile of CE -> LDS, with hardware padding
// of 4 dwords every 256 dwords => LDS row stride 520 bf16 (matches BSTRIDE).
__device__ __forceinline__ void tdm_load_tile(const unsigned short* gsrc, unsigned lds_off) {
    unsigned long long ga = (unsigned long long)(size_t)gsrc;
    u32x4 g0;
    g0[0] = 1u;                                        // count=1 (valid), user mode
    g0[1] = lds_off;                                   // lds_addr (bytes)
    g0[2] = (unsigned)(ga & 0xffffffffu);              // global_addr[31:0]
    g0[3] = (unsigned)((ga >> 32) & 0x01ffffffu)       // global_addr[56:32]
          | (2u << 30);                                // type = 2 ("image")
    i32x8 g1;
    g1[0] = (int)((1u << 16)      // data_size = 2 bytes
                | (1u << 20)      // pad_enable
                | (7u << 22)      // pad_interval: 256 dwords (=1024B row)
                | (3u << 25));    // pad_amount: 4 dwords (=8 bf16 -> stride 520)
    g1[1] = (int)(512u << 16);                         // tensor_dim0 = 512 (low 16)
    g1[2] = (int)(((unsigned)M_CAND & 0xffffu) << 16); // tensor_dim1 low 16
    g1[3] = (int)(((unsigned)M_CAND >> 16) | (512u << 16)); // tensor_dim1 hi | tile_dim0=512
    g1[4] = 16;                                        // tile_dim1 = 16 rows
    g1[5] = 512;                                       // tensor_dim0_stride = 512 elems
    g1[6] = 0;
    g1[7] = 0;
    i32x4 z4 = {0, 0, 0, 0};                           // groups 2/3 unused (2-D tensor)
    i32x8 z8 = {0, 0, 0, 0, 0, 0, 0, 0};
    __builtin_amdgcn_tensor_load_to_lds(g0, g1, z4, z4, z8, 0);   // 6-arg (clang-23) form
}
#endif

// ---------------- kernel 0: W[256x512] f32 -> Wt[512x256] bf16 ----------------
__global__ __launch_bounds__(256) void k_prep(const float* __restrict__ W,
                                              unsigned short* __restrict__ Wt) {
    int idx = blockIdx.x * 256 + threadIdx.x;
    int k = idx >> 9;
    int n = idx & 511;
    Wt[(size_t)n * D_IN + k] = f32_to_bf16(W[idx]);
}

// ---------------- kernel 1: E = inp @ W + b (bf16 out), WMMA ----------------
__global__ __launch_bounds__(256) void k_encode(const float* __restrict__ inp,
                                                const unsigned short* __restrict__ Wt,
                                                const float* __restrict__ bias,
                                                unsigned short* __restrict__ E) {
    const int lane = threadIdx.x & 31;
    const int wave = threadIdx.x >> 5;
    const int mrow = lane & 15;
    const int ha   = (lane >> 4) << 3;
    const int hb   = (lane >> 4) << 4;
    const int hr   = (lane >> 4) << 3;
    const int row_base = blockIdx.x * 16;

    const float* xrow = inp + (size_t)(row_base + mrow) * D_IN;

    BfVec a[8];
#pragma unroll
    for (int s = 0; s < 8; ++s) {
        const float* p0 = xrow + 32 * s + ha;
        float4 q0 = *(const float4*)(p0);
        float4 q1 = *(const float4*)(p0 + 4);
        float4 q2 = *(const float4*)(p0 + 16);
        float4 q3 = *(const float4*)(p0 + 20);
        a[s].u[0] = pk_bf16(q0.x, q0.y); a[s].u[1] = pk_bf16(q0.z, q0.w);
        a[s].u[2] = pk_bf16(q1.x, q1.y); a[s].u[3] = pk_bf16(q1.z, q1.w);
        a[s].u[4] = pk_bf16(q2.x, q2.y); a[s].u[5] = pk_bf16(q2.z, q2.w);
        a[s].u[6] = pk_bf16(q3.x, q3.y); a[s].u[7] = pk_bf16(q3.z, q3.w);
    }

#pragma unroll
    for (int ct = 0; ct < 4; ++ct) {
        const int cb = (wave + ct * 8) * 16;
        const int n  = cb + mrow;
        const unsigned short* wrow = Wt + (size_t)n * D_IN;
        v8f acc = {};
#pragma unroll
        for (int s = 0; s < 8; ++s) {
            BfVec bv;
            bv.q[0] = *(const uint4*)(wrow + 32 * s + ha);
            bv.q[1] = *(const uint4*)(wrow + 32 * s + 16 + ha);
            acc = wmma_bf16(a[s].v, bv.v, acc);
        }
        const float bn = bias[n];
#pragma unroll
        for (int v = 0; v < 8; ++v) {
            int row = row_base + v + hr;
            E[(size_t)row * DIM + n] = f32_to_bf16(acc[v] + bn);
        }
    }
}

// ---------------- kernel 2: row squared norms of bf16 matrix ----------------
__global__ __launch_bounds__(256) void k_norms(const unsigned short* __restrict__ E,
                                               float* __restrict__ out, int rows) {
    int wg   = (blockIdx.x * 256 + threadIdx.x) >> 5;
    int lane = threadIdx.x & 31;
    if (wg >= rows) return;
    const unsigned short* row = E + (size_t)wg * DIM;
    uint4 q0 = *(const uint4*)(row + lane * 16);
    uint4 q1 = *(const uint4*)(row + lane * 16 + 8);
    float s = 0.f;
    const unsigned int* u = (const unsigned int*)&q0;
#pragma unroll
    for (int i = 0; i < 4; ++i) { float a = bf16_lo(u[i]), b = bf16_hi(u[i]); s += a * a + b * b; }
    u = (const unsigned int*)&q1;
#pragma unroll
    for (int i = 0; i < 4; ++i) { float a = bf16_lo(u[i]), b = bf16_hi(u[i]); s += a * a + b * b; }
#pragma unroll
    for (int m = 16; m >= 1; m >>= 1) s += __shfl_xor(s, m, 32);
    if (lane == 0) out[wg] = s;
}

// ---------------- kernel 3: fused distance + exp + class accumulation ----------------
// grid = (RBLKS=16, NCHUNK=50); block = 256 (8 waves x 16 rows = 128 rows)
__global__ __launch_bounds__(256) void k_dist_fused(const unsigned short* __restrict__ XE,
                                                    const unsigned short* __restrict__ CE,
                                                    const float* __restrict__ X2,
                                                    const float* __restrict__ C2,
                                                    const int* __restrict__ CY,
                                                    float* __restrict__ pZ,
                                                    float* __restrict__ pAcc) {
#if USE_TDM
    __shared__ unsigned short Bs[2][16 * BSTRIDE];     // double-buffered TDM target
#else
    __shared__ unsigned short Bs[1][16 * BSTRIDE];
#endif
    __shared__ float x2s[RPB];
    __shared__ float Zs[RPB];
    __shared__ float As[RPB * D_OUT];

    const int tid  = threadIdx.x;
    const int lane = tid & 31;
    const int wave = tid >> 5;
    const int mrow = lane & 15;
    const int ha   = (lane >> 4) << 3;
    const int hb   = (lane >> 4) << 4;
    const int hr   = (lane >> 4) << 3;

    const int row_base   = blockIdx.x * RPB;
    const int chunk_base = blockIdx.y * CHUNK;

    if (tid < RPB) { x2s[tid] = X2[row_base + tid]; Zs[tid] = 0.f; }
    for (int i = tid; i < RPB * D_OUT; i += 256) As[i] = 0.f;

    // A tile for this wave, in registers across the whole M loop (16 K-steps)
    const unsigned short* arow = XE + (size_t)(row_base + wave * 16 + mrow) * DIM;
    BfVec areg[16];
#pragma unroll
    for (int s = 0; s < 16; ++s) {
        areg[s].q[0] = *(const uint4*)(arow + 32 * s + ha);
        areg[s].q[1] = *(const uint4*)(arow + 32 * s + 16 + ha);
    }

#if USE_TDM
    if (wave == 0)   // prime the pipeline: TDM load of tile 0 into buffer 0
        tdm_load_tile(CE + (size_t)chunk_base * DIM, (unsigned)(size_t)&Bs[0][0]);
#endif
    __syncthreads();

    int buf = 0;
    for (int t = 0; t < TILES; ++t) {
        const int cand_base = chunk_base + t * 16;
#if USE_TDM
        if (wave == 0) {
            if (t + 1 < TILES) {
                tdm_load_tile(CE + (size_t)(cand_base + 16) * DIM,
                              (unsigned)(size_t)&Bs[buf ^ 1][0]);
                __builtin_amdgcn_s_wait_tensorcnt(1);  // tile t finished; t+1 in flight
            } else {
                __builtin_amdgcn_s_wait_tensorcnt(0);
            }
        }
        __syncthreads();
#else
        for (int i = tid; i < 16 * 64; i += 256) {
            int r = i >> 6;
            int c = (i & 63) << 3;
            *(uint4*)&Bs[0][r * BSTRIDE + c] =
                *(const uint4*)(CE + (size_t)(cand_base + r) * DIM + c);
            if (t + 1 < TILES)
                __builtin_prefetch(CE + (size_t)(cand_base + 16 + r) * DIM + c, 0, 1);
        }
        __syncthreads();
#endif

        v8f acc = {};
#pragma unroll
        for (int s = 0; s < 16; ++s) {
            BfVec bv;
            bv.q[0] = *(const uint4*)(&Bs[buf][mrow * BSTRIDE + 32 * s + ha]);
            bv.q[1] = *(const uint4*)(&Bs[buf][mrow * BSTRIDE + 32 * s + 16 + ha]);
            acc = wmma_bf16(areg[s].v, bv.v, acc);
        }

        const int   cand = cand_base + mrow;
        const float c2n  = C2[cand];
        const int   cls  = CY[cand];
#pragma unroll
        for (int v = 0; v < 8; ++v) {
            const int   rl = wave * 16 + v + hr;
            const float d2 = x2s[rl] + c2n - 2.0f * acc[v];
            const float p  = __expf(-__fsqrt_rn(fmaxf(d2, 0.0f)));
            atomicAdd(&As[rl * D_OUT + cls], p);
            float ps = p;
            ps += __shfl_xor(ps, 1, 32);
            ps += __shfl_xor(ps, 2, 32);
            ps += __shfl_xor(ps, 4, 32);
            ps += __shfl_xor(ps, 8, 32);
            if (mrow == 0) atomicAdd(&Zs[rl], ps);
        }
        __syncthreads();
#if USE_TDM
        buf ^= 1;
#endif
    }

    float* pz = pZ + (size_t)blockIdx.y * N_ROWS + row_base;
    if (tid < RPB) pz[tid] = Zs[tid];
    float* pa = pAcc + ((size_t)blockIdx.y * N_ROWS + row_base) * D_OUT;
    for (int i = tid; i < RPB * D_OUT; i += 256) pa[i] = As[i];
}

// ---------------- kernel 4: reduce chunks, log(acc/Z + eps) ----------------
__global__ __launch_bounds__(256) void k_final(const float* __restrict__ pZ,
                                               const float* __restrict__ pAcc,
                                               float* __restrict__ out) {
    int row = blockIdx.x * 256 + threadIdx.x;
    if (row >= N_ROWS) return;
    float Z = 0.f;
    float a[D_OUT];
#pragma unroll
    for (int d = 0; d < D_OUT; ++d) a[d] = 0.f;
    for (int c = 0; c < NCHUNK; ++c) {
        Z += pZ[(size_t)c * N_ROWS + row];
        const float* pa = pAcc + ((size_t)c * N_ROWS + row) * D_OUT;
#pragma unroll
        for (int d = 0; d < D_OUT; ++d) a[d] += pa[d];
    }
    const float invZ = 1.0f / Z;
#pragma unroll
    for (int d = 0; d < D_OUT; ++d)
        out[(size_t)row * D_OUT + d] = __logf(a[d] * invZ + EPSV);
}

// ---------------- host ----------------
extern "C" void kernel_launch(void* const* d_in, const int* in_sizes, int n_in,
                              void* d_out, int out_size, void* d_ws, size_t ws_size,
                              hipStream_t stream) {
    const float* x  = (const float*)d_in[0];
    const float* cx = (const float*)d_in[2];
    const int*   cy = (const int*)d_in[3];
    const float* W  = (const float*)d_in[4];
    const float* b  = (const float*)d_in[5];
    float* out = (float*)d_out;

    char* ws = (char*)d_ws;
    unsigned short* Wt = (unsigned short*)(ws + OFF_WT);
    unsigned short* XE = (unsigned short*)(ws + OFF_XE);
    unsigned short* CE = (unsigned short*)(ws + OFF_CE);
    float* X2   = (float*)(ws + OFF_X2);
    float* C2   = (float*)(ws + OFF_C2);
    float* PZ   = (float*)(ws + OFF_PZ);
    float* PACC = (float*)(ws + OFF_PACC);

    k_prep<<<(D_IN * DIM) / 256, 256, 0, stream>>>(W, Wt);

    k_encode<<<N_ROWS / 16, 256, 0, stream>>>(x, Wt, b, XE);
    k_encode<<<M_CAND / 16, 256, 0, stream>>>(cx, Wt, b, CE);

    k_norms<<<N_ROWS / 8, 256, 0, stream>>>(XE, X2, N_ROWS);
    k_norms<<<M_CAND / 8, 256, 0, stream>>>(CE, C2, M_CAND);

    dim3 g3(RBLKS, NCHUNK);
    k_dist_fused<<<g3, 256, 0, stream>>>(XE, CE, X2, C2, cy, PZ, PACC);

    k_final<<<(N_ROWS + 255) / 256, 256, 0, stream>>>(PZ, PACC, out);
}
